// Model_17995912970498
// MI455X (gfx1250) — matrix operand order repaired and checked by reading, not compiled, not run on hardware
//
#include <hip/hip_runtime.h>

// ---------------------------------------------------------------------------
// MI455X (gfx1250): ResNet18 backbone as implicit-GEMM bf16 WMMA convolutions
// (v_wmma_f32_16x16x32_bf16), capsule einsum as f16 WMMA GEMM, routing as
// wave32 reductions. Compute-bound (~116 GFLOP) => WMMA path.
// This revision: LDS double buffering in both WMMA kernels (register prefetch
// of K-step k+32 overlaps global loads with WMMA math; one barrier per
// K-step), branchless 8B-chunk A staging in the capsule GEMM.
// ---------------------------------------------------------------------------

typedef __bf16    bf16_t;
typedef _Float16  f16_t;
typedef __attribute__((ext_vector_type(16))) __bf16   v16bf;
typedef __attribute__((ext_vector_type(8)))  __bf16   v8bf;
typedef __attribute__((ext_vector_type(16))) _Float16 v16h;
typedef __attribute__((ext_vector_type(8)))  _Float16 v8h;
typedef __attribute__((ext_vector_type(8)))  float    v8f;
typedef __attribute__((ext_vector_type(4)))  unsigned v4u;
typedef __attribute__((ext_vector_type(2)))  unsigned v2u;

// ======================= prep kernels ======================================

__global__ void k_bnprep(const float* __restrict__ g, const float* __restrict__ be,
                         const float* __restrict__ mu, const float* __restrict__ va,
                         float* __restrict__ scale, float* __restrict__ bias, int C) {
  int i = blockIdx.x * 256 + threadIdx.x;
  if (i < C) {
    float s = g[i] * rsqrtf(va[i] + 1e-5f);
    scale[i] = s;
    bias[i]  = be[i] - mu[i] * s;
  }
}

// OIHW f32 -> [CO][Kc] bf16 with k = (kh*KW+kw)*CIt + ci, zero padded.
__global__ void k_packw(const float* __restrict__ w, bf16_t* __restrict__ wp,
                        int CO, int CIr, int CIt, int KH, int KW, int Kc) {
  int i = blockIdx.x * 256 + threadIdx.x;
  int tot = CO * Kc;
  if (i >= tot) return;
  int co = i / Kc;
  int k  = i - co * Kc;
  float v = 0.f;
  int K = KH * KW * CIt;
  if (k < K) {
    int tap = k / CIt;
    int ci  = k - tap * CIt;
    if (ci < CIr) {
      int kh = tap / KW, kw = tap - (tap / KW) * KW;
      v = w[(((long)co * CIr + ci) * KH + kh) * KW + kw];
    }
  }
  wp[i] = (bf16_t)v;
}

// x: [32,3,224,224] f32 NCHW  ->  [32,224,224,4] bf16 NHWC (4th channel zero)
__global__ void k_in_cvt(const float* __restrict__ x, bf16_t* __restrict__ y) {
  const int HW = 224 * 224;
  int i = blockIdx.x * 256 + threadIdx.x;
  int tot = 32 * HW;
  if (i >= tot) return;
  int b = i / HW, p = i - b * HW;
  bf16_t* o = y + (long)i * 4;
  o[0] = (bf16_t)x[((long)b * 3 + 0) * HW + p];
  o[1] = (bf16_t)x[((long)b * 3 + 1) * HW + p];
  o[2] = (bf16_t)x[((long)b * 3 + 2) * HW + p];
  o[3] = (bf16_t)0.f;
}

// 3x3 s2 p1 maxpool on NHWC bf16: [32,112,112,64] -> [32,56,56,64]
__global__ void k_maxpool(const bf16_t* __restrict__ in, bf16_t* __restrict__ out) {
  int i = blockIdx.x * 256 + threadIdx.x;
  int tot = 32 * 56 * 56 * 64;
  if (i >= tot) return;
  int c = i & 63;
  int t = i >> 6;
  int ow = t % 56; t /= 56;
  int oh = t % 56;
  int b  = t / 56;
  float m = -3.4e38f;
  for (int kh = 0; kh < 3; ++kh) {
    int ih = oh * 2 + kh - 1;
    if (ih < 0 || ih >= 112) continue;
    for (int kw = 0; kw < 3; ++kw) {
      int iw = ow * 2 + kw - 1;
      if (iw < 0 || iw >= 112) continue;
      m = fmaxf(m, (float)in[(((long)b * 112 + ih) * 112 + iw) * 64 + c]);
    }
  }
  out[i] = (bf16_t)m;
}

__global__ void k_w16(const float* __restrict__ w, f16_t* __restrict__ o, int n) {
  int i = blockIdx.x * 256 + threadIdx.x;
  if (i < n) o[i] = (f16_t)w[i];
}

// ======================= implicit-GEMM conv (bf16 WMMA) ====================
// M = 32*HO*WO output pixels, N = CO, K = KH*KW*CI (CI power of 2, even).
// Block: 256 threads = 8 waves (4x2), tile 128(M) x 64(N), BK=32.
// Double-buffered LDS: register-prefetch K-step k+32 while WMMAs consume the
// current buffer; one barrier per K-step.
// Fused: y = relu?( conv*scale + bias [+ residual] ).
// CIBIG: CI % 32 == 0 -> each 16B K-chunk stays inside one filter tap.
// kwMagic = ceil(256/KW): kh = (tap*kwMagic)>>8 exact for tap <= 55.

template <bool CIBIG>
__global__ __launch_bounds__(256) void k_conv_bf16_wmma(
    const bf16_t* __restrict__ in, const bf16_t* __restrict__ wp,
    const float* __restrict__ scale, const float* __restrict__ bias,
    const bf16_t* __restrict__ resid, bf16_t* __restrict__ out,
    int M, int IH, int IW, int CI, int ciShift, int HO, int WO, int CO,
    int KW, int kwMagic, int stride, int pad, int K, int Kc, int relu) {
  __shared__ alignas(16) bf16_t Asm[2][128 * 40];  // [buf][m][k], stride 40
  __shared__ alignas(16) bf16_t Bsm[2][64 * 40];   // [buf][n][k]

  const int tid  = threadIdx.x;
  const int lane = tid & 31;
  const int wv   = tid >> 5;
  const int wm   = wv >> 1;          // 0..3
  const int wn   = wv & 1;           // 0..1
  const int m0   = blockIdx.x * 128;
  const int n0   = blockIdx.y * 64;
  const int HW   = HO * WO;
  const int lm   = lane & 15;
  const int kbA  = (lane < 16) ? 0 : 8;
  const int kbB  = (lane < 16) ? 0 : 8;

  // hoisted per-row output coordinates for this thread's A-staging rows
  constexpr int NT = CIBIG ? 2 : 8;
  int  r_ohs[NT], r_ows[NT], r_ok[NT];
  long r_base[NT];
#pragma unroll
  for (int t = 0; t < NT; ++t) {
    int idx = tid + t * 256;
    int row = CIBIG ? (idx >> 2) : (idx >> 4);
    int m   = m0 + row;
    r_ok[t] = (m < M);
    int mm  = r_ok[t] ? m : 0;
    int b   = mm / HW;
    int rm  = mm - b * HW;
    int oh  = rm / WO;
    int ow  = rm - oh * WO;
    r_ohs[t]  = oh * stride - pad;
    r_ows[t]  = ow * stride - pad;
    r_base[t] = (long)b * IH;
  }

  // ---- branchless prefetch of one K-step into registers ----
  v4u ra4[2], rb4;               // CIBIG path
  unsigned ra1[8], rb1[4];       // stem path
  auto fetch = [&](int k0) {
    if constexpr (CIBIG) {
#pragma unroll
      for (int t = 0; t < 2; ++t) {
        int idx = tid + t * 256;
        int ch  = idx & 3;
        int k   = k0 + ch * 8;
        int tap = k >> ciShift;
        int ci  = k & (CI - 1);
        int kh  = (tap * kwMagic) >> 8;
        int kw  = tap - kh * KW;
        int ih  = r_ohs[t] + kh;
        int iw  = r_ows[t] + kw;
        bool inb = r_ok[t] && (ih >= 0) && (ih < IH) && (iw >= 0) && (iw < IW);
        long off = ((r_base[t] + ih) * IW + iw) * (long)CI + ci;
        long offc = inb ? off : 0;
        v4u val = *(const v4u*)(in + offc);
        if (!inb) { v4u z = {0u, 0u, 0u, 0u}; val = z; }
        ra4[t] = val;
      }
      rb4 = *(const v4u*)(wp + (long)(n0 + (tid >> 2)) * Kc + k0 + (tid & 3) * 8);
    } else {
#pragma unroll
      for (int t = 0; t < 8; ++t) {
        int idx = tid + t * 256;
        int kp  = idx & 15;
        int k   = k0 + kp * 2;
        int tap = k >> ciShift;
        int ci  = k & (CI - 1);
        int kh  = (tap * kwMagic) >> 8;
        int kw  = tap - kh * KW;
        int ih  = r_ohs[t] + kh;
        int iw  = r_ows[t] + kw;
        bool inb = r_ok[t] && (k < K) &&
                   (ih >= 0) && (ih < IH) && (iw >= 0) && (iw < IW);
        long off = ((r_base[t] + ih) * IW + iw) * (long)CI + ci;
        long offc = inb ? off : 0;
        unsigned val = *(const unsigned*)(in + offc);
        if (!inb) val = 0u;
        ra1[t] = val;
      }
#pragma unroll
      for (int t = 0; t < 4; ++t) {
        int idx = tid + t * 256;
        rb1[t] = *(const unsigned*)(wp + (long)(n0 + (idx >> 4)) * Kc + k0 + (idx & 15) * 2);
      }
    }
  };
  auto store = [&](int buf) {
    if constexpr (CIBIG) {
#pragma unroll
      for (int t = 0; t < 2; ++t) {
        int idx = tid + t * 256;
        *(v4u*)(&Asm[buf][(idx >> 2) * 40 + (idx & 3) * 8]) = ra4[t];
      }
      *(v4u*)(&Bsm[buf][(tid >> 2) * 40 + (tid & 3) * 8]) = rb4;
    } else {
#pragma unroll
      for (int t = 0; t < 8; ++t) {
        int idx = tid + t * 256;
        *(unsigned*)(&Asm[buf][(idx >> 4) * 40 + (idx & 15) * 2]) = ra1[t];
      }
#pragma unroll
      for (int t = 0; t < 4; ++t) {
        int idx = tid + t * 256;
        *(unsigned*)(&Bsm[buf][(idx >> 4) * 40 + (idx & 15) * 2]) = rb1[t];
      }
    }
  };

  v8f acc[2][2] = {};

  fetch(0);
  store(0);
  int cur = 0;
  for (int k0 = 0; k0 < Kc; k0 += 32) {
    __syncthreads();                       // buf[cur] ready for all waves
    bool hasNext = (k0 + 32) < Kc;
    if (hasNext) fetch(k0 + 32);           // overlaps WMMA below

    // ---- fragment loads per CDNA5 16-bit A/B VGPR layouts ----
    v16bf af[2], bfr[2];
#pragma unroll
    for (int tm = 0; tm < 2; ++tm) {
      const bf16_t* pa = &Asm[cur][(wm * 32 + tm * 16 + lm) * 40 + kbA];
      v8bf lo = *(const v8bf*)pa;           // K kbA..kbA+7
      v8bf hi = *(const v8bf*)(pa + 16);    // K kbA+16..kbA+23
      af[tm] = __builtin_shufflevector(lo, hi, 0,1,2,3,4,5,6,7,8,9,10,11,12,13,14,15);
    }
#pragma unroll
    for (int tn = 0; tn < 2; ++tn) {
      const bf16_t* pb = &Bsm[cur][(wn * 32 + tn * 16 + lm) * 40 + kbB];
      v8bf lo = *(const v8bf*)pb;           // K kbB..kbB+7
      v8bf hi = *(const v8bf*)(pb + 16);    // K kbB+16..kbB+23
      bfr[tn] = __builtin_shufflevector(lo, hi, 0,1,2,3,4,5,6,7,8,9,10,11,12,13,14,15);
    }
#pragma unroll
    for (int tm = 0; tm < 2; ++tm)
#pragma unroll
      for (int tn = 0; tn < 2; ++tn)
        acc[tm][tn] = __builtin_amdgcn_wmma_f32_16x16x32_bf16(
            false, af[tm], false, bfr[tn], (short)0, acc[tm][tn], false, false);

    if (hasNext) store(cur ^ 1);           // next iteration's barrier orders this
    cur ^= 1;
  }

  // ---- epilogue: BN + residual + relu, C layout: M=r+8*(lane>=16), N=lane&15
  const int mhalf = (lane >> 4) << 3;
#pragma unroll
  for (int tm = 0; tm < 2; ++tm) {
#pragma unroll
    for (int tn = 0; tn < 2; ++tn) {
      int co = n0 + wn * 32 + tn * 16 + lm;
      float sc = scale[co], bi = bias[co];
#pragma unroll
      for (int r = 0; r < 8; ++r) {
        int m = m0 + wm * 32 + tm * 16 + r + mhalf;
        if (m < M) {
          float v = acc[tm][tn][r] * sc + bi;
          long off = (long)m * CO + co;
          if (resid) v += (float)resid[off];
          if (relu)  v = fmaxf(v, 0.f);
          out[off] = (bf16_t)v;
        }
      }
    }
  }
}

// ======================= capsule einsum (f16 WMMA) =========================
// priors[b,i,u] = sum_v W[i,u,v] * hidden[b,i,v].
// Per capsule i: GEMM M=32(b) x N=512(u) x K=512(v). Block tile 32x64, 8 waves
// 2x4, double-buffered LDS, branchless 8B A-chunks (one per thread).

__global__ __launch_bounds__(256) void k_capsule_gemm(
    const f16_t* __restrict__ hid,   // [32][8][512]
    const f16_t* __restrict__ wf,    // [8][512][512]  (u-major, contiguous v)
    float* __restrict__ priors) {    // [32][8][512]
  __shared__ alignas(16) f16_t Ash[2][32 * 40];
  __shared__ alignas(16) f16_t Bsh[2][64 * 40];
  const int tid  = threadIdx.x;
  const int lane = tid & 31;
  const int wv   = tid >> 5;
  const int wm   = wv >> 2;     // 0..1
  const int wn   = wv & 3;      // 0..3
  const int i    = blockIdx.y;
  const int n0   = blockIdx.x * 64;
  const int lm   = lane & 15;
  const int kbA  = (lane < 16) ? 0 : 8;
  const int kbB  = (lane < 16) ? 0 : 8;
  // A: 256 8B chunks (32 rows x 8), one per thread (exec-uniform)
  const int arow = tid >> 3, ach = tid & 7;
  // B: 256 16B chunks (64 rows x 4), one per thread
  const int brow = tid >> 2, bch = tid & 3;

  v2u raf; v4u rbf;
  auto fetch = [&](int k0) {
    raf = *(const v2u*)(hid + ((long)arow * 8 + i) * 512 + k0 + ach * 4);
    rbf = *(const v4u*)(wf + ((long)i * 512 + n0 + brow) * 512 + k0 + bch * 8);
  };
  auto store = [&](int buf) {
    *(v2u*)(&Ash[buf][arow * 40 + ach * 4]) = raf;
    *(v4u*)(&Bsh[buf][brow * 40 + bch * 8]) = rbf;
  };

  v8f acc = {};
  fetch(0);
  store(0);
  int cur = 0;
  for (int k0 = 0; k0 < 512; k0 += 32) {
    __syncthreads();
    bool hasNext = (k0 + 32) < 512;
    if (hasNext) fetch(k0 + 32);

    const f16_t* pa = &Ash[cur][(wm * 16 + lm) * 40 + kbA];
    v8h alo = *(const v8h*)pa;
    v8h ahi = *(const v8h*)(pa + 16);
    v16h a = __builtin_shufflevector(alo, ahi, 0,1,2,3,4,5,6,7,8,9,10,11,12,13,14,15);
    const f16_t* pb = &Bsh[cur][(wn * 16 + lm) * 40 + kbB];
    v8h blo = *(const v8h*)pb;
    v8h bhi = *(const v8h*)(pb + 16);
    v16h b = __builtin_shufflevector(blo, bhi, 0,1,2,3,4,5,6,7,8,9,10,11,12,13,14,15);

    acc = __builtin_amdgcn_wmma_f32_16x16x32_f16(false, a, false, b, (short)0, acc, false, false);

    if (hasNext) store(cur ^ 1);
    cur ^= 1;
  }
  const int u = n0 + wn * 16 + lm;
  const int mhalf = (lane >> 4) << 3;
#pragma unroll
  for (int r = 0; r < 8; ++r) {
    int m = wm * 16 + r + mhalf;
    priors[((long)m * 8 + i) * 512 + u] = acc[r];
  }
}

// ======================= k-means routing ===================================
// Stage 1: per (b,k): x[p,c] = att[k]*feat[b,p,c], p<49, c<512; 3 iterations.
__global__ __launch_bounds__(256) void k_routing1(
    const bf16_t* __restrict__ feat,   // [32][49][512] NHWC final features
    const float*  __restrict__ att,    // [8]
    float* __restrict__ hidden,        // [32][8][512]
    f16_t* __restrict__ hidden16) {
  const int b  = blockIdx.x >> 3;
  const int kc = blockIdx.x & 7;
  const int tid = threadIdx.x, lane = tid & 31, wv = tid >> 5;
  const float a = att[kc];
  const bf16_t* xb = feat + (long)b * 49 * 512;

  __shared__ float v[512];
  __shared__ float red[256];
  __shared__ float pr[49];
  __shared__ float sc[2];

  // v = mean over 49 positions
  float s0 = 0.f, s1 = 0.f;
  for (int p = 0; p < 49; ++p) {
    s0 += (float)xb[p * 512 + tid];
    s1 += (float)xb[p * 512 + tid + 256];
  }
  v[tid]       = a * s0 * (1.f / 49.f);
  v[tid + 256] = a * s1 * (1.f / 49.f);
  __syncthreads();

  for (int it = 0; it < 3; ++it) {
    // 1/max(||v||, eps)
    red[tid] = v[tid] * v[tid] + v[tid + 256] * v[tid + 256];
    __syncthreads();
    for (int st = 128; st > 0; st >>= 1) {
      if (tid < st) red[tid] += red[tid + st];
      __syncthreads();
    }
    if (tid == 0) sc[0] = 1.f / fmaxf(sqrtf(red[0]), 1e-12f);
    __syncthreads();
    float inv = sc[0];

    // logits[p] = <x_p, v/||v||>  (wave per row, wave32 shuffle reduction)
    for (int p = wv; p < 49; p += 8) {
      float s = 0.f;
      for (int c = lane; c < 512; c += 32) s += a * (float)xb[p * 512 + c] * v[c];
      for (int o = 16; o > 0; o >>= 1) s += __shfl_xor(s, o, 32);
      if (lane == 0) pr[p] = s * inv;
    }
    __syncthreads();
    // softmax over 49 positions
    if (tid == 0) {
      float mx = -3.4e38f;
      for (int p = 0; p < 49; ++p) mx = fmaxf(mx, pr[p]);
      float ss = 0.f;
      for (int p = 0; p < 49; ++p) { float e = __expf(pr[p] - mx); pr[p] = e; ss += e; }
      sc[1] = 1.f / ss;
    }
    __syncthreads();
    float isum = sc[1];
    float n0 = 0.f, n1 = 0.f;
    for (int p = 0; p < 49; ++p) {
      float pp = pr[p];
      n0 += pp * (float)xb[p * 512 + tid];
      n1 += pp * (float)xb[p * 512 + tid + 256];
    }
    n0 *= a * isum;
    n1 *= a * isum;
    __syncthreads();
    v[tid] = n0; v[tid + 256] = n1;
    __syncthreads();
    // squash
    red[tid] = n0 * n0 + n1 * n1;
    __syncthreads();
    for (int st = 128; st > 0; st >>= 1) {
      if (tid < st) red[tid] += red[tid + st];
      __syncthreads();
    }
    if (tid == 0) { float sq = red[0]; sc[0] = (sq / (1.f + sq)) / sqrtf(sq); }
    __syncthreads();
    float f = sc[0];
    v[tid] *= f; v[tid + 256] *= f;
    __syncthreads();
  }
  long o = ((long)b * 8 + kc) * 512;
  hidden[o + tid]         = v[tid];
  hidden[o + tid + 256]   = v[tid + 256];
  hidden16[o + tid]       = (f16_t)v[tid];
  hidden16[o + tid + 256] = (f16_t)v[tid + 256];
}

// Stage 2: per b: route 8 capsule priors (1 iteration) then L2-normalize.
__global__ __launch_bounds__(256) void k_routing2(
    const float* __restrict__ priors,  // [32][8][512]
    float* __restrict__ outp) {        // [32][512]
  const int b = blockIdx.x;
  const int tid = threadIdx.x, lane = tid & 31, wv = tid >> 5;
  const float* xb = priors + (long)b * 8 * 512;
  __shared__ float v[512];
  __shared__ float red[256];
  __shared__ float pr[8];
  __shared__ float sc[2];

  float s0 = 0.f, s1 = 0.f;
  for (int j = 0; j < 8; ++j) {
    s0 += xb[j * 512 + tid];
    s1 += xb[j * 512 + tid + 256];
  }
  v[tid] = s0 * 0.125f; v[tid + 256] = s1 * 0.125f;
  __syncthreads();

  red[tid] = v[tid] * v[tid] + v[tid + 256] * v[tid + 256];
  __syncthreads();
  for (int st = 128; st > 0; st >>= 1) { if (tid < st) red[tid] += red[tid + st]; __syncthreads(); }
  if (tid == 0) sc[0] = 1.f / fmaxf(sqrtf(red[0]), 1e-12f);
  __syncthreads();
  float inv = sc[0];
  {
    int j = wv;                         // 8 waves -> 8 capsules
    float s = 0.f;
    for (int c = lane; c < 512; c += 32) s += xb[j * 512 + c] * v[c];
    for (int o = 16; o > 0; o >>= 1) s += __shfl_xor(s, o, 32);
    if (lane == 0) pr[j] = s * inv;
  }
  __syncthreads();
  if (tid == 0) {
    float mx = -3.4e38f;
    for (int j = 0; j < 8; ++j) mx = fmaxf(mx, pr[j]);
    float ss = 0.f;
    for (int j = 0; j < 8; ++j) { float e = __expf(pr[j] - mx); pr[j] = e; ss += e; }
    sc[1] = 1.f / ss;
  }
  __syncthreads();
  float isum = sc[1];
  float n0 = 0.f, n1 = 0.f;
  for (int j = 0; j < 8; ++j) {
    float pp = pr[j];
    n0 += pp * xb[j * 512 + tid];
    n1 += pp * xb[j * 512 + tid + 256];
  }
  n0 *= isum; n1 *= isum;
  __syncthreads();
  v[tid] = n0; v[tid + 256] = n1;
  __syncthreads();
  red[tid] = n0 * n0 + n1 * n1;
  __syncthreads();
  for (int st = 128; st > 0; st >>= 1) { if (tid < st) red[tid] += red[tid + st]; __syncthreads(); }
  if (tid == 0) {
    float sq = red[0];
    float f  = (sq / (1.f + sq)) / sqrtf(sq);   // squash factor
    float nn = f * sqrtf(sq);                   // ||squash(v)||
    sc[0] = f;
    sc[1] = 1.f / fmaxf(nn, 1e-12f);
  }
  __syncthreads();
  float fo = sc[0] * sc[1];
  outp[(long)b * 512 + tid]       = v[tid] * fo;
  outp[(long)b * 512 + tid + 256] = v[tid + 256] * fo;
}

// ======================= host orchestration ================================

extern "C" void kernel_launch(void* const* d_in, const int* in_sizes, int n_in,
                              void* d_out, int out_size, void* d_ws, size_t ws_size,
                              hipStream_t stream) {
  (void)in_sizes; (void)n_in; (void)out_size; (void)ws_size;
  static const int CIN[8]  = {64, 64, 64, 128, 128, 256, 256, 512};
  static const int COUT[8] = {64, 64, 128, 128, 256, 256, 512, 512};

  // ---- parse inputs (depth-first insertion order of setup_inputs dict) ----
  int idx = 0;
  const float* x_f32  = (const float*)d_in[idx++];
  const float* stem_w = (const float*)d_in[idx++];
  const float* stem_bn[4];
  for (int j = 0; j < 4; ++j) stem_bn[j] = (const float*)d_in[idx++];
  const float *c1[8], *bn1[8][4], *c2[8], *bn2[8][4], *dw[8], *dbn[8][4];
  for (int i = 0; i < 8; ++i) {
    c1[i] = (const float*)d_in[idx++];
    for (int j = 0; j < 4; ++j) bn1[i][j] = (const float*)d_in[idx++];
    c2[i] = (const float*)d_in[idx++];
    for (int j = 0; j < 4; ++j) bn2[i][j] = (const float*)d_in[idx++];
    if (CIN[i] != COUT[i]) {
      dw[i] = (const float*)d_in[idx++];
      for (int j = 0; j < 4; ++j) dbn[i][j] = (const float*)d_in[idx++];
    } else {
      dw[i] = nullptr;
      for (int j = 0; j < 4; ++j) dbn[i][j] = nullptr;
    }
  }
  const float* att  = (const float*)d_in[idx++];
  const float* refw = (const float*)d_in[idx++];

  // ---- workspace carve ----
  char* wsp = (char*)d_ws;
  auto alloc = [&](size_t bytes) -> void* {
    void* r = (void*)wsp;
    wsp += (bytes + 255) & ~(size_t)255;
    return r;
  };
  bf16_t* x4       = (bf16_t*)alloc((size_t)32 * 224 * 224 * 4 * 2);
  bf16_t* stem_out = (bf16_t*)alloc((size_t)32 * 112 * 112 * 64 * 2);
  bf16_t* bufs[4];
  for (int j = 0; j < 4; ++j) bufs[j] = (bf16_t*)alloc((size_t)32 * 56 * 56 * 64 * 2);
  bf16_t* stem_wp = (bf16_t*)alloc((size_t)64 * 224 * 2);
  float*  stem_s  = (float*)alloc(64 * 4);
  float*  stem_b  = (float*)alloc(64 * 4);
  bf16_t *wp1[8], *wp2[8], *wpd[8];
  float  *s1[8], *b1[8], *s2[8], *b2[8], *sd[8], *bd[8];
  int Kc1[8], Kc2[8], Kcd[8];
  for (int i = 0; i < 8; ++i) {
    Kc1[i] = 9 * CIN[i];                         // multiples of 32 for CI>=64
    Kc2[i] = 9 * COUT[i];
    wp1[i] = (bf16_t*)alloc((size_t)COUT[i] * Kc1[i] * 2);
    wp2[i] = (bf16_t*)alloc((size_t)COUT[i] * Kc2[i] * 2);
    s1[i] = (float*)alloc(COUT[i] * 4); b1[i] = (float*)alloc(COUT[i] * 4);
    s2[i] = (float*)alloc(COUT[i] * 4); b2[i] = (float*)alloc(COUT[i] * 4);
    if (dw[i]) {
      Kcd[i] = CIN[i];
      wpd[i] = (bf16_t*)alloc((size_t)COUT[i] * Kcd[i] * 2);
      sd[i] = (float*)alloc(COUT[i] * 4); bd[i] = (float*)alloc(COUT[i] * 4);
    } else { Kcd[i] = 0; wpd[i] = nullptr; sd[i] = bd[i] = nullptr; }
  }
  float* hidden   = (float*)alloc((size_t)32 * 8 * 512 * 4);
  f16_t* hidden16 = (f16_t*)alloc((size_t)32 * 8 * 512 * 2);
  f16_t* w16      = (f16_t*)alloc((size_t)8 * 512 * 512 * 2);
  float* priors   = (float*)alloc((size_t)32 * 8 * 512 * 4);

  // ---- prep: conversions, weight packing, BN folding ----
  k_in_cvt<<<(32 * 224 * 224 + 255) / 256, 256, 0, stream>>>(x_f32, x4);
  k_packw<<<(64 * 224 + 255) / 256, 256, 0, stream>>>(stem_w, stem_wp, 64, 3, 4, 7, 7, 224);
  k_bnprep<<<1, 256, 0, stream>>>(stem_bn[0], stem_bn[1], stem_bn[2], stem_bn[3], stem_s, stem_b, 64);
  for (int i = 0; i < 8; ++i) {
    k_packw<<<((COUT[i] * Kc1[i]) + 255) / 256, 256, 0, stream>>>(c1[i], wp1[i], COUT[i], CIN[i], CIN[i], 3, 3, Kc1[i]);
    k_packw<<<((COUT[i] * Kc2[i]) + 255) / 256, 256, 0, stream>>>(c2[i], wp2[i], COUT[i], COUT[i], COUT[i], 3, 3, Kc2[i]);
    k_bnprep<<<(COUT[i] + 255) / 256, 256, 0, stream>>>(bn1[i][0], bn1[i][1], bn1[i][2], bn1[i][3], s1[i], b1[i], COUT[i]);
    k_bnprep<<<(COUT[i] + 255) / 256, 256, 0, stream>>>(bn2[i][0], bn2[i][1], bn2[i][2], bn2[i][3], s2[i], b2[i], COUT[i]);
    if (dw[i]) {
      k_packw<<<((COUT[i] * Kcd[i]) + 255) / 256, 256, 0, stream>>>(dw[i], wpd[i], COUT[i], CIN[i], CIN[i], 1, 1, Kcd[i]);
      k_bnprep<<<(COUT[i] + 255) / 256, 256, 0, stream>>>(dbn[i][0], dbn[i][1], dbn[i][2], dbn[i][3], sd[i], bd[i], COUT[i]);
    }
  }
  k_w16<<<(8 * 512 * 512 + 255) / 256, 256, 0, stream>>>(refw, w16, 8 * 512 * 512);

  auto conv = [&](const bf16_t* in, const bf16_t* wpk, const float* sc, const float* bi,
                  const bf16_t* res, bf16_t* out, int IH, int IW, int CI,
                  int HO, int WO, int CO, int KH, int KW, int st, int pd, int relu) {
    int K = KH * KW * CI;
    int Kc = (K + 31) & ~31;
    int M = 32 * HO * WO;
    int ciShift = __builtin_ctz(CI);
    int kwMagic = (255 + KW) / KW;        // ceil(256/KW): exact kh=(tap*m)>>8
    dim3 g((M + 127) / 128, CO / 64);
    if ((CI & 31) == 0) {
      k_conv_bf16_wmma<true><<<g, 256, 0, stream>>>(in, wpk, sc, bi, res, out,
                                                    M, IH, IW, CI, ciShift, HO, WO, CO,
                                                    KW, kwMagic, st, pd, K, Kc, relu);
    } else {
      k_conv_bf16_wmma<false><<<g, 256, 0, stream>>>(in, wpk, sc, bi, res, out,
                                                     M, IH, IW, CI, ciShift, HO, WO, CO,
                                                     KW, kwMagic, st, pd, K, Kc, relu);
    }
  };

  // ---- stem: 7x7 s2 p3 (CI padded to 4, K=196 -> Kc=224) + maxpool ----
  conv(x4, stem_wp, stem_s, stem_b, nullptr, stem_out, 224, 224, 4, 112, 112, 64, 7, 7, 2, 3, 1);
  k_maxpool<<<(32 * 56 * 56 * 64 + 255) / 256, 256, 0, stream>>>(stem_out, bufs[0]);

  // ---- residual blocks with buffer rotation ----
  int a = 0, H = 56;
  for (int i = 0; i < 8; ++i) {
    int ci_ = CIN[i], co_ = COUT[i];
    int s   = (ci_ != co_) ? 2 : 1;
    int HOd = H / s;
    bf16_t* h   = bufs[a];
    bf16_t* y   = bufs[(a + 1) & 3];
    bf16_t* scb = bufs[(a + 2) & 3];
    bf16_t* ob  = bufs[(a + 3) & 3];
    conv(h, wp1[i], s1[i], b1[i], nullptr, y, H, H, ci_, HOd, HOd, co_, 3, 3, s, 1, 1);
    const bf16_t* res = h;
    if (s == 2) {
      conv(h, wpd[i], sd[i], bd[i], nullptr, scb, H, H, ci_, HOd, HOd, co_, 1, 1, 2, 0, 0);
      res = scb;
    }
    conv(y, wp2[i], s2[i], b2[i], res, ob, HOd, HOd, co_, HOd, HOd, co_, 3, 3, 1, 1, 1);
    a = (a + 3) & 3;
    H = HOd;
  }

  // ---- capsule routing + einsum + final routing/normalize ----
  k_routing1<<<256, 256, 0, stream>>>(bufs[a], att, hidden, hidden16);
  k_capsule_gemm<<<dim3(8, 8), 256, 0, stream>>>(hidden16, w16, priors);
  k_routing2<<<32, 256, 0, stream>>>(priors, (float*)d_out);
}
